// EvoformerBlock_32959579029862
// MI455X (gfx1250) — matrix operand order repaired and checked by reading, not compiled, not run on hardware
//
#include <hip/hip_runtime.h>
#include <hip/hip_bf16.h>
#include <math.h>

// ---------------------------------------------------------------------------
// Evoformer block for MI455X (gfx1250): all matmuls via v_wmma_f32_16x16x32_bf16
// ---------------------------------------------------------------------------

typedef __attribute__((ext_vector_type(8)))  float  v8f;
typedef __attribute__((ext_vector_type(16))) __bf16 v16bf;

#define NRES 256
#define NN   65536   // NRES*NRES
#define INFV 1e9f

#if __has_builtin(__builtin_amdgcn_wave_barrier)
#define WAVE_BAR() __builtin_amdgcn_wave_barrier()
#else
#define WAVE_BAR() asm volatile("" ::: "memory")
#endif

// ---------------------------------------------------------------------------
// LayerNorm over last dim. blockDim.x == C (128 or 256). One block per row.
// trans!=0: read source row (j*Nside+i) for logical row (i*Nside+j).
// ---------------------------------------------------------------------------
__global__ __launch_bounds__(256) void ln_kernel(
    const float* __restrict__ x, const float* __restrict__ g,
    const float* __restrict__ b, float* __restrict__ y,
    int C, int Nside, int trans) {
  int row = blockIdx.x;
  long long src = row;
  if (trans) { int i = row / Nside, j = row - i * Nside; src = (long long)j * Nside + i; }
  int c = threadIdx.x;
  float v = x[src * C + c];
  __shared__ float red[8];
  float s = v;
#pragma unroll
  for (int o = 16; o >= 1; o >>= 1) s += __shfl_xor(s, o, 32);
  if ((threadIdx.x & 31) == 0) red[threadIdx.x >> 5] = s;
  __syncthreads();
  int nw = blockDim.x >> 5;
  float tot = 0.f;
  for (int w = 0; w < nw; ++w) tot += red[w];
  float mu = tot / (float)C;
  __syncthreads();
  float d = v - mu;
  s = d * d;
#pragma unroll
  for (int o = 16; o >= 1; o >>= 1) s += __shfl_xor(s, o, 32);
  if ((threadIdx.x & 31) == 0) red[threadIdx.x >> 5] = s;
  __syncthreads();
  float tv = 0.f;
  for (int w = 0; w < nw; ++w) tv += red[w];
  float var = tv / (float)C;
  y[(long long)row * C + c] = d * rsqrtf(var + 1e-5f) * g[c] + b[c];
}

// ---------------------------------------------------------------------------
// Generic strided batched GEMM: C = epi(A @ B + bias). bf16 WMMA, fp32 acc.
// A element (m,k) at A + z*Ab + m*Asm + k*Ask
// B element (k,n) at B + z*Bb + k*Bsk + n*Bsn
// C element (m,n) at C + z*Cb + m*Csm + n*Csn
// Block: 256 threads = 8 waves; wave w computes 16x16 tile (row mt, col nt).
// EPI: 0 = none, 1 = +bias, 2 = sigmoid(+bias)
// ---------------------------------------------------------------------------
template <int EPI>
__global__ __launch_bounds__(256) void gemm_wmma_kernel(
    const float* __restrict__ A, const float* __restrict__ B,
    const float* __restrict__ bias, float* __restrict__ C,
    int M, int Nc, int K,
    long long Asm, long long Ask, long long Ab,
    long long Bsk, long long Bsn, long long Bb,
    long long Csm, long long Csn, long long Cb) {
  A += (long long)blockIdx.z * Ab;
  B += (long long)blockIdx.z * Bb;
  C += (long long)blockIdx.z * Cb;
  int mt = blockIdx.x;
  int wid = threadIdx.x >> 5, lane = threadIdx.x & 31;
  int l16 = lane & 15, lhi = lane >> 4;
  int nt = blockIdx.y * 8 + wid;
  bool active = (nt * 16) < Nc;
  int ncl = nt * 16 + l16; if (ncl >= Nc) ncl = Nc - 1;   // clamped gather col

  __shared__ __bf16 sA[16][32];
  v8f acc = {0.f, 0.f, 0.f, 0.f, 0.f, 0.f, 0.f, 0.f};

  for (int kb = 0; kb < K; kb += 32) {
    // cooperative stage of the 16x32 A tile (bf16)
    {
      int e = threadIdx.x;
      int r = e >> 5, kk = e & 31;
      sA[r][kk] = (__bf16)A[(long long)(mt * 16 + r) * Asm + (long long)(kb + kk) * Ask];
      e += 256; r = e >> 5; kk = e & 31;
      sA[r][kk] = (__bf16)A[(long long)(mt * 16 + r) * Asm + (long long)(kb + kk) * Ask];
    }
    __syncthreads();
    if (active) {
      v16bf av, bv;
#pragma unroll
      for (int e = 0; e < 16; ++e) {
        av[e] = sA[l16][(e < 8 ? e : e + 8) + lhi * 8];                         // A operand layout
        bv[e] = (__bf16)B[(long long)(kb + (e < 8 ? e : e + 8) + lhi * 8) * Bsk + (long long)ncl * Bsn]; // B operand layout
      }
      if (kb + 32 < K)  // prefetch next B k-block (global_prefetch_b8)
        __builtin_prefetch(&B[(long long)(kb + 32 + lhi * 16) * Bsk + (long long)ncl * Bsn], 0, 1);
      acc = __builtin_amdgcn_wmma_f32_16x16x32_bf16(false, av, false, bv, (short)0, acc, false, false);
    }
    __syncthreads();
  }
  if (active && (nt * 16 + l16) < Nc) {
#pragma unroll
    for (int r = 0; r < 8; ++r) {
      int m = mt * 16 + r + lhi * 8;
      int n = nt * 16 + l16;
      float v = acc[r];
      if (EPI >= 1) v += bias[n];
      if (EPI == 2) v = 1.f / (1.f + __expf(-v));
      C[(long long)m * Csm + (long long)n * Csn] = v;
    }
  }
}

// ---------------------------------------------------------------------------
// Fused attention with online softmax (flash style), d = 32.
// logits(i,h,j,k) = scale*Q.K + INF*(mask(i,k)-1) + TB(j,k,h); O = softmax @ V,
// then O *= G (gate already sigmoided), stored via so* strides.
// Grid: (Jrows/128, H, I); block 256 = 8 waves; one 16-row j-tile per wave.
// Running max/sum are register-resident (replicated across each 16-lane column
// group); only the P layout shuffle (C-layout -> A-operand) goes through LDS.
// ---------------------------------------------------------------------------
__global__ __launch_bounds__(256) void fused_att_kernel(
    const float* __restrict__ Q, const float* __restrict__ Kp,
    const float* __restrict__ V, const float* __restrict__ G,
    const float* __restrict__ TB, const float* __restrict__ MSK,
    float* __restrict__ O, int Ktot,
    long long sqi, long long sqj, long long sqh,
    long long ski, long long skk, long long skh,
    long long svi, long long svk, long long svh,
    long long sgi, long long sgj, long long sgh,
    long long stj, long long stk, long long sth,
    long long smi, long long smk,
    long long soi, long long soj, long long soh,
    float scale) {
  int i = blockIdx.z, h = blockIdx.y;
  int wid = threadIdx.x >> 5, lane = threadIdx.x & 31;
  int l16 = lane & 15, lhi = lane >> 4;
  int jt = blockIdx.x * 8 + wid;

  __shared__ __bf16 sP[8][16][32];

  float rm[8], rs[8];
#pragma unroll
  for (int r = 0; r < 8; ++r) { rm[r] = -1e30f; rs[r] = 0.f; }

  // Q operand is reused for all K chunks (d == WMMA K-dim == 32)
  int jlane = jt * 16 + l16;
  v16bf qa;
#pragma unroll
  for (int e = 0; e < 16; ++e) {
    int d = (e < 8 ? e : e + 8) + lhi * 8;
    qa[e] = (__bf16)Q[(long long)i * sqi + (long long)jlane * sqj + (long long)h * sqh + d];
  }

  v8f acc0 = {0.f, 0.f, 0.f, 0.f, 0.f, 0.f, 0.f, 0.f};
  v8f acc1 = {0.f, 0.f, 0.f, 0.f, 0.f, 0.f, 0.f, 0.f};

  for (int kc = 0; kc < Ktot; kc += 32) {
    // ---- logits: two 16x16 WMMAs (K^T operand, reduction over d) ----
    v16bf kb0, kb1;
#pragma unroll
    for (int e = 0; e < 16; ++e) {
      int d = (e < 8 ? e : e + 8) + lhi * 8;
      kb0[e] = (__bf16)Kp[(long long)i * ski + (long long)(kc + l16) * skk + (long long)h * skh + d];
      kb1[e] = (__bf16)Kp[(long long)i * ski + (long long)(kc + 16 + l16) * skk + (long long)h * skh + d];
    }
    v8f s0 = {0.f, 0.f, 0.f, 0.f, 0.f, 0.f, 0.f, 0.f};
    v8f s1 = {0.f, 0.f, 0.f, 0.f, 0.f, 0.f, 0.f, 0.f};
    s0 = __builtin_amdgcn_wmma_f32_16x16x32_bf16(false, qa, false, kb0, (short)0, s0, false, false);
    s1 = __builtin_amdgcn_wmma_f32_16x16x32_bf16(false, qa, false, kb1, (short)0, s1, false, false);

    float mt0 = INFV * (MSK[(long long)i * smi + (long long)(kc + l16) * smk] - 1.f);
    float mt1 = INFV * (MSK[(long long)i * smi + (long long)(kc + 16 + l16) * smk] - 1.f);

    // ---- per-row online softmax, register-resident running stats ----
#pragma unroll
    for (int r = 0; r < 8; ++r) {
      int jr = jt * 16 + r + 8 * lhi;
      float v0 = s0[r] * scale + mt0 + TB[(long long)jr * stj + (long long)(kc + l16) * stk + (long long)h * sth];
      float v1 = s1[r] * scale + mt1 + TB[(long long)jr * stj + (long long)(kc + 16 + l16) * stk + (long long)h * sth];
      float mx = fmaxf(v0, v1);
#pragma unroll
      for (int o = 8; o >= 1; o >>= 1) mx = fmaxf(mx, __shfl_xor(mx, o, 32));  // row max (16-lane group)
      float nm = fmaxf(rm[r], mx);
      float f  = __expf(rm[r] - nm);
      rm[r] = nm;
      float p0 = __expf(v0 - nm), p1 = __expf(v1 - nm);
      float ps = p0 + p1;
#pragma unroll
      for (int o = 8; o >= 1; o >>= 1) ps += __shfl_xor(ps, o, 32);            // row sum
      rs[r] = rs[r] * f + ps;
      acc0[r] *= f; acc1[r] *= f;
      sP[wid][r + 8 * lhi][l16] = (__bf16)p0;
      sP[wid][r + 8 * lhi][16 + l16] = (__bf16)p1;
    }
    WAVE_BAR();

    // ---- O += P @ V : P as bf16 A operand from LDS, V as B operand ----
    v16bf pa;
#pragma unroll
    for (int e = 0; e < 16; ++e)
      pa[e] = sP[wid][l16][(e < 8 ? e : e + 8) + lhi * 8];
    v16bf vb0, vb1;
#pragma unroll
    for (int e = 0; e < 16; ++e) {
      long long kk = kc + (e < 8 ? e : e + 8) + lhi * 8;
      vb0[e] = (__bf16)V[(long long)i * svi + kk * svk + (long long)h * svh + l16];
      vb1[e] = (__bf16)V[(long long)i * svi + kk * svk + (long long)h * svh + 16 + l16];
    }
    acc0 = __builtin_amdgcn_wmma_f32_16x16x32_bf16(false, pa, false, vb0, (short)0, acc0, false, false);
    acc1 = __builtin_amdgcn_wmma_f32_16x16x32_bf16(false, pa, false, vb1, (short)0, acc1, false, false);
    WAVE_BAR();
  }

  // ---- finalize: normalize, gate, store ----
#pragma unroll
  for (int r = 0; r < 8; ++r) {
    int jr = jt * 16 + r + 8 * lhi;
    float inv = 1.0f / rs[r];
    int d0 = l16, d1 = 16 + l16;
    float o0 = acc0[r] * inv * G[(long long)i * sgi + (long long)jr * sgj + (long long)h * sgh + d0];
    float o1 = acc1[r] * inv * G[(long long)i * sgi + (long long)jr * sgj + (long long)h * sgh + d1];
    O[(long long)i * soi + (long long)jr * soj + (long long)h * soh + d0] = o0;
    O[(long long)i * soi + (long long)jr * soj + (long long)h * soh + d1] = o1;
  }
}

// ---------------------------------------------------------------------------
// Fused transition: out[row,:] += mask[row] * (relu(xl @ W1 + b1) @ W2 + b2)
// Hidden (16 x 4C, bf16) kept in LDS. Block: 256 threads, 16 rows per block.
// ---------------------------------------------------------------------------
template <int CC>
__global__ __launch_bounds__(256) void transition_kernel(
    const float* __restrict__ xl,
    const float* __restrict__ w1, const float* __restrict__ b1,
    const float* __restrict__ w2, const float* __restrict__ b2,
    const float* __restrict__ mask, float* __restrict__ out, int M) {
  const int HID = 4 * CC;
  __shared__ __bf16 sX[16][CC];
  __shared__ __bf16 sH[16][HID];
  int mt = blockIdx.x;
  int wid = threadIdx.x >> 5, lane = threadIdx.x & 31;
  int l16 = lane & 15, lhi = lane >> 4;

  for (int e = threadIdx.x; e < 16 * CC; e += 256) {
    int r = e / CC, c = e - r * CC;
    sX[r][c] = (__bf16)xl[(long long)(mt * 16 + r) * CC + c];
  }
  __syncthreads();

  // phase 1: hidden = relu(xl @ W1 + b1) -> LDS (bf16)
  const int HT = HID / (16 * 8);
#pragma unroll
  for (int ht = 0; ht < HT; ++ht) {
    int hcol = (wid * HT + ht) * 16;
    v8f acc = {0.f, 0.f, 0.f, 0.f, 0.f, 0.f, 0.f, 0.f};
    for (int kb = 0; kb < CC; kb += 32) {
      v16bf av, bv;
#pragma unroll
      for (int e = 0; e < 16; ++e) {
        av[e] = sX[l16][kb + (e < 8 ? e : e + 8) + lhi * 8];
        bv[e] = (__bf16)w1[(long long)(kb + (e < 8 ? e : e + 8) + lhi * 8) * HID + hcol + l16];
      }
      acc = __builtin_amdgcn_wmma_f32_16x16x32_bf16(false, av, false, bv, (short)0, acc, false, false);
    }
#pragma unroll
    for (int r = 0; r < 8; ++r) {
      int row = r + 8 * lhi, col = hcol + l16;
      float v = acc[r] + b1[col];
      sH[row][col] = (__bf16)(v > 0.f ? v : 0.f);
    }
  }
  __syncthreads();

  // phase 2: out += mask * (hidden @ W2 + b2)
  const int OT = CC / 128;
#pragma unroll
  for (int ot = 0; ot < OT; ++ot) {
    int ocol = (wid * OT + ot) * 16;
    v8f acc = {0.f, 0.f, 0.f, 0.f, 0.f, 0.f, 0.f, 0.f};
    for (int kb = 0; kb < HID; kb += 32) {
      v16bf av, bv;
#pragma unroll
      for (int e = 0; e < 16; ++e) {
        av[e] = sH[l16][kb + (e < 8 ? e : e + 8) + lhi * 8];
        bv[e] = (__bf16)w2[(long long)(kb + (e < 8 ? e : e + 8) + lhi * 8) * CC + ocol + l16];
      }
      acc = __builtin_amdgcn_wmma_f32_16x16x32_bf16(false, av, false, bv, (short)0, acc, false, false);
    }
#pragma unroll
    for (int r = 0; r < 8; ++r) {
      int rowg = mt * 16 + r + 8 * lhi, col = ocol + l16;
      float v = (acc[r] + b2[col]) * mask[rowg];
      out[(long long)rowg * CC + col] += v;
    }
  }
}

// ---------------------------------------------------------------------------
// Pointwise helpers (float4 vectorized; C is a power of two, passed as shift)
// ---------------------------------------------------------------------------
__global__ __launch_bounds__(256) void mask_gate_kernel(
    float4* __restrict__ p, const float4* __restrict__ g,
    const float* __restrict__ msk, long long total4, int Cshift) {
  long long e = (long long)blockIdx.x * 256 + threadIdx.x;
  if (e >= total4) return;
  float mk = msk[(e * 4) >> Cshift];
  float4 gv = g[e], pv = p[e];
  pv.x = mk * gv.x * pv.x; pv.y = mk * gv.y * pv.y;
  pv.z = mk * gv.z * pv.z; pv.w = mk * gv.w * pv.w;
  p[e] = pv;
}

__global__ __launch_bounds__(256) void resid_add_kernel(
    float* __restrict__ z, const float4* __restrict__ t,
    const float4* __restrict__ g, long long total4, int Cshift,
    int Nside, int trans) {
  long long e = (long long)blockIdx.x * 256 + threadIdx.x;
  if (e >= total4) return;
  float4 v = t[e];
  if (g) { float4 gv = g[e]; v.x *= gv.x; v.y *= gv.y; v.z *= gv.z; v.w *= gv.w; }
  long long base = e * 4;
  long long dst = base;
  if (trans) {
    long long row = base >> Cshift;
    int c = (int)(base & ((1LL << Cshift) - 1));
    long long i = row / Nside, j = row - i * Nside;
    dst = ((j * Nside + i) << Cshift) + c;
  }
  float4* zp = (float4*)(z + dst);
  float4 zv = *zp;
  zv.x += v.x; zv.y += v.y; zv.z += v.z; zv.w += v.w;
  *zp = zv;
}

// ---------------------------------------------------------------------------
// Host orchestration
// Input flat order (setup_inputs, depth-first dict order):
//  0:m 1:z 2:single_mask 3:pair_mask
//  tmo@4, tmi@20 (16 each): ln_in.g, ln_in.b, a_p_w, a_p_b, a_g_w, a_g_b,
//                           b_p_w, b_p_b, b_g_w, b_g_b, ln_out.g, ln_out.b,
//                           z_w, z_b, g_w, g_b
//  tas@36, tae@46 (10 each): ln.g, ln.b, bias_w, q_w, k_w, v_w, g_w, g_b, o_w, o_b
//  pt@56 (6): ln.g, ln.b, w1, b1, w2, b2
//  sa@62 (12): ln_m.g, ln_m.b, ln_z.g, ln_z.b, zb_w, q_w, k_w, v_w, g_w, g_b, o_w, o_b
//  st@74 (6): ln.g, ln.b, w1, b1, w2, b2
// Output: m (65536) then z (8388608), concatenated.
// ---------------------------------------------------------------------------
extern "C" void kernel_launch(void* const* d_in, const int* in_sizes, int n_in,
                              void* d_out, int out_size, void* d_ws, size_t ws_size,
                              hipStream_t stream) {
  (void)in_sizes; (void)n_in; (void)out_size; (void)ws_size;
  const int N = NRES;
  const float* m_in  = (const float*)d_in[0];
  const float* z_in  = (const float*)d_in[1];
  const float* smask = (const float*)d_in[2];
  const float* pmask = (const float*)d_in[3];
  auto P = [&](int i) { return (const float*)d_in[i]; };

  float* mO = (float*)d_out;         // working m  [256,256]
  float* zO = mO + NN;               // working z  [256,256,128]

  float* ws = (float*)d_ws;
  const size_t BB = (size_t)NN * 128;
  float* W0 = ws;            // zl
  float* W1 = ws + BB;       // a / q / proj-out
  float* W2 = ws + 2 * BB;   // b / k / xl
  float* W3 = ws + 3 * BB;   // gates / v
  float* W4 = ws + 4 * BB;   // x / attention gate
  float* TBb = ws + 5 * BB;  // pair bias, up to NN*8
  float* ml  = TBb + (size_t)NN * 8;
  float* sq  = ml  + NN;
  float* sk_ = sq  + NN;
  float* sv  = sk_ + NN;
  float* sg  = sv  + NN;
  float* so  = sg  + NN;

  hipMemcpyAsync(mO, m_in, (size_t)NN * sizeof(float), hipMemcpyDeviceToDevice, stream);
  hipMemcpyAsync(zO, z_in, BB * sizeof(float), hipMemcpyDeviceToDevice, stream);

  auto ln = [&](const float* x, const float* g, const float* b, float* y, int R, int C, int trans) {
    ln_kernel<<<R, C, 0, stream>>>(x, g, b, y, C, N, trans);
  };
  auto gemm = [&](int epi, const float* A, const float* Bm, const float* bias, float* Cm,
                  int M, int Nc, int K,
                  long long Asm, long long Ask, long long Ab,
                  long long Bsk, long long Bsn, long long Bb,
                  long long Csm, long long Csn, long long Cb, int batch) {
    dim3 g(M / 16, (Nc + 127) / 128, batch);
    dim3 blk(256);
    if (epi == 0)
      gemm_wmma_kernel<0><<<g, blk, 0, stream>>>(A, Bm, bias, Cm, M, Nc, K, Asm, Ask, Ab, Bsk, Bsn, Bb, Csm, Csn, Cb);
    else if (epi == 1)
      gemm_wmma_kernel<1><<<g, blk, 0, stream>>>(A, Bm, bias, Cm, M, Nc, K, Asm, Ask, Ab, Bsk, Bsn, Bb, Csm, Csn, Cb);
    else
      gemm_wmma_kernel<2><<<g, blk, 0, stream>>>(A, Bm, bias, Cm, M, Nc, K, Asm, Ask, Ab, Bsk, Bsn, Bb, Csm, Csn, Cb);
  };
  auto proj128 = [&](int epi, const float* A, const float* Wm, const float* bias, float* Cm) {
    gemm(epi, A, Wm, bias, Cm, NN, 128, 128, 128, 1, 0, 128, 1, 0, 128, 1, 0, 1);
  };
  const long long PW  = (long long)NN * 128;
  const long long PW4 = PW / 4;
  dim3 pwg((unsigned)((PW4 + 255) / 256));

  // ---------------- TriangleMultiplication (outgoing / incoming) ----------
  auto tri_mul = [&](int ib, bool outgoing) {
    ln(zO, P(ib + 0), P(ib + 1), W0, NN, 128, 0);
    proj128(2, W0, P(ib + 4), P(ib + 5), W3);                                    // sigmoid(zl@a_g_w+b)
    proj128(1, W0, P(ib + 2), P(ib + 3), W1);                                    // zl@a_p_w+b
    mask_gate_kernel<<<pwg, 256, 0, stream>>>((float4*)W1, (const float4*)W3, pmask, PW4, 7);  // a
    proj128(2, W0, P(ib + 8), P(ib + 9), W3);                                    // sigmoid(zl@b_g_w+b)
    proj128(1, W0, P(ib + 6), P(ib + 7), W2);                                    // zl@b_p_w+b
    mask_gate_kernel<<<pwg, 256, 0, stream>>>((float4*)W2, (const float4*)W3, pmask, PW4, 7);  // b
    long long sNC = (long long)N * 128;
    if (outgoing)  // x[i,j,c] = sum_k a[i,k,c] b[j,k,c]
      gemm(0, W1, W2, nullptr, W4, N, N, N, sNC, 128, 1, 128, sNC, 1, sNC, 128, 1, 128);
    else           // x[i,j,c] = sum_k a[k,i,c] b[k,j,c]
      gemm(0, W1, W2, nullptr, W4, N, N, N, 128, sNC, 1, sNC, 128, 1, sNC, 128, 1, 128);
    ln(W4, P(ib + 10), P(ib + 11), W2, NN, 128, 0);                              // LN(x)
    proj128(2, W0, P(ib + 14), P(ib + 15), W3);                                  // sigmoid(zl@g_w+b)
    proj128(1, W2, P(ib + 12), P(ib + 13), W1);                                  // LN(x)@z_w+b
    resid_add_kernel<<<pwg, 256, 0, stream>>>(zO, (const float4*)W1, (const float4*)W3, PW4, 7, N, 0);
  };

  // ---------------- TriangleAttention (start / end via transpose) ---------
  auto tri_att = [&](int ib, bool trans) {
    ln(zO, P(ib + 0), P(ib + 1), W0, NN, 128, trans ? 1 : 0);
    proj128(0, W0, P(ib + 3), nullptr, W1);                                      // q
    proj128(0, W0, P(ib + 4), nullptr, W2);                                      // k
    proj128(0, W0, P(ib + 5), nullptr, W3);                                      // v
    proj128(2, W0, P(ib + 6), P(ib + 7), W4);                                    // gate = sigmoid(.)
    gemm(0, W0, P(ib + 2), nullptr, TBb, NN, 4, 128, 128, 1, 0, 4, 1, 0, 4, 1, 0, 1); // pair bias tb[q,k,h]
    long long s = (long long)N * 128;
    dim3 ga(2, 4, N);
    fused_att_kernel<<<ga, 256, 0, stream>>>(
        W1, W2, W3, W4, TBb, pmask, W0, N,
        s, 128, 32,  s, 128, 32,  s, 128, 32,  s, 128, 32,
        (long long)N * 4, 4, 1,
        trans ? 1LL : (long long)N, trans ? (long long)N : 1LL,
        s, 128, 32, 0.17677669529663687f);
    proj128(1, W0, P(ib + 8), P(ib + 9), W1);                                    // o@o_w+o_b
    resid_add_kernel<<<pwg, 256, 0, stream>>>(zO, (const float4*)W1, nullptr, PW4, 7, N, trans ? 1 : 0);
  };

  tri_mul(4, true);    // tmo
  tri_mul(20, false);  // tmi
  tri_att(36, false);  // tas (starting node)
  tri_att(46, true);   // tae (ending node, transposed)

  // ---------------- Pair transition ---------------------------------------
  ln(zO, P(56), P(57), W0, NN, 128, 0);
  transition_kernel<128><<<NN / 16, 256, 0, stream>>>(W0, P(58), P(59), P(60), P(61), pmask, zO, NN);

  // ---------------- Single attention with pair bias -----------------------
  {
    int ib = 62;
    ln(mO, P(ib + 0), P(ib + 1), ml, N, 256, 0);
    ln(zO, P(ib + 2), P(ib + 3), W0, NN, 128, 0);
    gemm(0, ml, P(ib + 5), nullptr, sq,  N, 256, 256, 256, 1, 0, 256, 1, 0, 256, 1, 0, 1);
    gemm(0, ml, P(ib + 6), nullptr, sk_, N, 256, 256, 256, 1, 0, 256, 1, 0, 256, 1, 0, 1);
    gemm(0, ml, P(ib + 7), nullptr, sv,  N, 256, 256, 256, 1, 0, 256, 1, 0, 256, 1, 0, 1);
    gemm(2, ml, P(ib + 8), P(ib + 9), sg, N, 256, 256, 256, 1, 0, 256, 1, 0, 256, 1, 0, 1);
    gemm(0, W0, P(ib + 4), nullptr, TBb, NN, 8, 128, 128, 1, 0, 8, 1, 0, 8, 1, 0, 1);  // zb[q,k,h]
    dim3 ga(2, 8, 1);
    fused_att_kernel<<<ga, 256, 0, stream>>>(
        sq, sk_, sv, sg, TBb, smask, so, N,
        0, 256, 32,  0, 256, 32,  0, 256, 32,  0, 256, 32,
        (long long)N * 8, 8, 1,
        0, 1,
        0, 256, 32, 0.17677669529663687f);
    gemm(1, so, P(ib + 10), P(ib + 11), sk_, N, 256, 256, 256, 1, 0, 256, 1, 0, 256, 1, 0, 1);
    resid_add_kernel<<<dim3((NN / 4 + 255) / 256), 256, 0, stream>>>(mO, (const float4*)sk_, nullptr, NN / 4, 8, N, 0);
  }

  // ---------------- Single transition -------------------------------------
  ln(mO, P(74), P(75), ml, N, 256, 0);
  transition_kernel<256><<<N / 16, 256, 0, stream>>>(ml, P(76), P(77), P(78), P(79), smask, mO, N);
}
